// SVDHead_32633161515825
// MI455X (gfx1250) — compile-verified, matrix operand order checked by reading, not run on hardware
//
#include <hip/hip_runtime.h>

#define BB 8
#define NN 1024
#define KK 32
#define PPK 256

typedef __attribute__((ext_vector_type(16))) _Float16 v16h;
typedef __attribute__((ext_vector_type(8)))  _Float16 v8h;
typedef __attribute__((ext_vector_type(8)))  float    v8f;

// ---------------- param index map (insertion order of params dict) ----------
enum {
  C1W = 0, C1B, BN1G, BN1B, BN1M, BN1V,
  C2W, C2B, BN2G, BN2B, BN2M, BN2V,
  M2C1W, M2C1B, M2BN1G, M2BN1B, M2BN1M, M2BN1V,
  M2C2W, M2C2B, M2BN2G, M2BN2B, M2BN2M, M2BN2V,
  M3C1W, M3C1B, M3BN1G, M3BN1B, M3BN1M, M3BN1V,
  M3C2W, M3C2B, M3BN2G, M3BN2B, M3BN2M, M3BN2V,
  M4C1W, M4C1B, M4BNG, M4BNB, M4BNM, M4BNV,
  M4C2W, M4C2B, NPARAMS
};

struct DiscParams { const float* p[NPARAMS]; };

// ---------------- helpers ---------------------------------------------------
__device__ __forceinline__ float lrelu(float x) { return x > 0.f ? x : 0.2f * x; }
__device__ __forceinline__ v8f wmma16(v16h a, v16h b, v8f c) {
  return __builtin_amdgcn_wmma_f32_16x16x32_f16(false, a, false, b, (short)0, c, false, false);
}
// Load one lane's 32-half contiguous K-slice as a WMMA fragment.
// ISA layout: frag elements [0..7] = K 8h..8h+7, [8..15] = K 16+8h..23+8h.
__device__ __forceinline__ v16h load_frag(const _Float16* slice, int hh) {
  v8h lo = *(const v8h*)(slice + (hh << 3));
  v8h hi = *(const v8h*)(slice + 16 + (hh << 3));
  return __builtin_shufflevector(lo, hi, 0, 1, 2, 3, 4, 5, 6, 7,
                                 8, 9, 10, 11, 12, 13, 14, 15);
}

// ---------------- K1: src_corr[b,d,m] = sum_n tgt[b,d,n]*scores[b,m,n] ------
__global__ __launch_bounds__(128)
void k_src_corr(const float* __restrict__ tgt, const float* __restrict__ scores,
                float* __restrict__ corr) {
  __shared__ float red[3][128];
  int b = blockIdx.x >> 10, m = blockIdx.x & 1023, tid = threadIdx.x;
  const float* srow = scores + (size_t)(b * NN + m) * NN;
  const float* t0 = tgt + (size_t)(b * 3 + 0) * NN;
  const float* t1 = tgt + (size_t)(b * 3 + 1) * NN;
  const float* t2 = tgt + (size_t)(b * 3 + 2) * NN;
  float a0 = 0.f, a1 = 0.f, a2 = 0.f;
  for (int n = tid; n < NN; n += 128) {
    float sc = srow[n];
    a0 += t0[n] * sc; a1 += t1[n] * sc; a2 += t2[n] * sc;
  }
  red[0][tid] = a0; red[1][tid] = a1; red[2][tid] = a2;
  __syncthreads();
  for (int s = 64; s > 0; s >>= 1) {
    if (tid < s) {
      red[0][tid] += red[0][tid + s];
      red[1][tid] += red[1][tid + s];
      red[2][tid] += red[2][tid + s];
    }
    __syncthreads();
  }
  if (tid == 0) {
    corr[(b * 3 + 0) * NN + m] = red[0][0];
    corr[(b * 3 + 1) * NN + m] = red[1][0];
    corr[(b * 3 + 2) * NN + m] = red[2][0];
  }
}

// ---------------- K2: fused discriminator -> conf[b,n] ----------------------
#define NFLOATS 1492
#define OFF_W1F   0
#define OFF_BE1   576
#define OFF_BE2   640
#define OFF_M2W2  768
#define OFF_M2BE1 1024
#define OFF_M2BE2 1040
#define OFF_M3W2  1056
#define OFF_M3BE1 1312
#define OFF_M3BE2 1328
#define OFF_M4W1  1344
#define OFF_M4BE1 1472
#define OFF_M4W2  1480
#define OFF_M4B2  1488
#define HOFF_W2H  0
#define HOFF_M2W1 24576
#define HOFF_M3W1 26624
#define NHALF     28672
#define PWBASE    (NFLOATS * 4 + NHALF * 2)   // 63312 (16B aligned)
#define PWSTRIDE  19200
#define DISC_LDS_BYTES (PWBASE + 4 * PWSTRIDE) // 140112

__device__ __forceinline__ void mlp16(const _Float16* __restrict__ Wh,
                                      const float* __restrict__ be1v,
                                      const float* __restrict__ W2,
                                      const float* __restrict__ be2v,
                                      const _Float16* __restrict__ x2t,
                                      float* __restrict__ t1,
                                      float* __restrict__ outb, int lane) {
  int m = lane & 15, hh = lane >> 4;
  v8f c0 = {}; v8f c1 = {};
#pragma unroll
  for (int st = 0; st < 4; ++st) {
    v16h af = load_frag(Wh + m * 128 + st * 32, hh);
    v16h b0 = load_frag(x2t + m * 128 + st * 32, hh);
    v16h b1 = load_frag(x2t + (16 + m) * 128 + st * 32, hh);
    c0 = wmma16(af, b0, c0);
    c1 = wmma16(af, b1, c1);
  }
#pragma unroll
  for (int r = 0; r < 8; ++r) {
    int o = r + (hh << 3);
    t1[o * 32 + m]      = lrelu(c0[r] + be1v[o]);
    t1[o * 32 + 16 + m] = lrelu(c1[r] + be1v[o]);
  }
  __syncthreads();
  for (int o = 0; o < 16; ++o) {
    float acc = be2v[o];
#pragma unroll
    for (int c = 0; c < 16; ++c) acc += W2[o * 16 + c] * t1[c * 32 + lane];
    outb[o * 32 + lane] = lrelu(acc);
  }
  __syncthreads();
}

__global__ __launch_bounds__(128)
void k_disc(DiscParams dp, const float* __restrict__ src,
            const float* __restrict__ src_knn, const int* __restrict__ src_idx,
            const float* __restrict__ corr, float* __restrict__ conf) {
  extern __shared__ char smem[];
  float*    F = (float*)smem;
  _Float16* H = (_Float16*)(smem + NFLOATS * 4);
  int tid = threadIdx.x;

  // -------- stage BN-folded weights (once per block) --------
  for (int i = tid; i < 576; i += 128) {
    int c = i / 9;
    float s = dp.p[BN1G][c] * rsqrtf(dp.p[BN1V][c] + 1e-5f);
    F[OFF_W1F + i] = dp.p[C1W][i] * s;
  }
  for (int c = tid; c < 64; c += 128) {
    float s = dp.p[BN1G][c] * rsqrtf(dp.p[BN1V][c] + 1e-5f);
    F[OFF_BE1 + c] = (dp.p[C1B][c] - dp.p[BN1M][c]) * s + dp.p[BN1B][c];
  }
  for (int i = tid; i < 128 * 192; i += 128) {
    int o = i / 192, r = i % 192, tau = r >> 6, ci = r & 63;
    float s = dp.p[BN2G][o] * rsqrtf(dp.p[BN2V][o] + 1e-5f);
    H[HOFF_W2H + i] = (_Float16)(dp.p[C2W][(o * 64 + ci) * 3 + tau] * s);
  }
  for (int o = tid; o < 128; o += 128) {
    float s = dp.p[BN2G][o] * rsqrtf(dp.p[BN2V][o] + 1e-5f);
    F[OFF_BE2 + o] = (dp.p[C2B][o] - dp.p[BN2M][o]) * s + dp.p[BN2B][o];
  }
  for (int g = 0; g < 2; ++g) {
    int base = (g == 0) ? M2C1W : M3C1W;
    int hw   = (g == 0) ? HOFF_M2W1 : HOFF_M3W1;
    int ow2  = (g == 0) ? OFF_M2W2 : OFF_M3W2;
    int ob1  = (g == 0) ? OFF_M2BE1 : OFF_M3BE1;
    int ob2  = (g == 0) ? OFF_M2BE2 : OFF_M3BE2;
    for (int i = tid; i < 2048; i += 128) {
      int o = i >> 7;
      float s = dp.p[base + 2][o] * rsqrtf(dp.p[base + 5][o] + 1e-5f);
      H[hw + i] = (_Float16)(dp.p[base][i] * s);
    }
    for (int o = tid; o < 16; o += 128) {
      float s = dp.p[base + 2][o] * rsqrtf(dp.p[base + 5][o] + 1e-5f);
      F[ob1 + o] = (dp.p[base + 1][o] - dp.p[base + 4][o]) * s + dp.p[base + 3][o];
    }
    for (int i = tid; i < 256; i += 128) {
      int o = i >> 4;
      float s = dp.p[base + 8][o] * rsqrtf(dp.p[base + 11][o] + 1e-5f);
      F[ow2 + i] = dp.p[base + 6][i] * s;
    }
    for (int o = tid; o < 16; o += 128) {
      float s = dp.p[base + 8][o] * rsqrtf(dp.p[base + 11][o] + 1e-5f);
      F[ob2 + o] = (dp.p[base + 7][o] - dp.p[base + 10][o]) * s + dp.p[base + 9][o];
    }
  }
  for (int i = tid; i < 128; i += 128) {
    int o = i >> 4;
    float s = dp.p[M4BNG][o] * rsqrtf(dp.p[M4BNV][o] + 1e-5f);
    F[OFF_M4W1 + i] = dp.p[M4C1W][i] * s;
  }
  if (tid < 8) {
    float s = dp.p[M4BNG][tid] * rsqrtf(dp.p[M4BNV][tid] + 1e-5f);
    F[OFF_M4BE1 + tid] = (dp.p[M4C1B][tid] - dp.p[M4BNM][tid]) * s + dp.p[M4BNB][tid];
    F[OFF_M4W2 + tid]  = dp.p[M4C2W][tid];
  }
  if (tid == 0) F[OFF_M4B2] = dp.p[M4C2B][0];
  __syncthreads();

  // -------- per-wave point processing --------
  int wave = tid >> 5, lane = tid & 31;
  char* wb = smem + PWBASE + wave * PWSTRIDE;
  float* xin  = (float*)wb;          // 104 floats (3x34 padded)
  float* t1   = xin + 104;           // 512
  float* a2b  = t1 + 512;            // 512
  float* l3b  = a2b + 512;           // 512
  float* sbuf = l3b + 512;           // 16
  float* h4b  = sbuf + 16;           // 8
  _Float16* h1t = (_Float16*)(wb + 6656); // [34][64] halves, kpos-major
  _Float16* x2t = h1t + 2176;             // [32][128] halves, kpos-major

  int pt = blockIdx.x * 4 + wave;        // b*N + n
  int b = pt >> 10, n = pt & 1023;
  int m = lane & 15, hh = lane >> 4;

#pragma unroll
  for (int br = 0; br < 2; ++br) {
    // build padded conv input [3][34] for this point (lane = kpos)
    if (br == 0) {
      int q = src_idx[pt * KK + lane];
      int b2 = q >> 10, n2 = q & 1023;
#pragma unroll
      for (int i = 0; i < 3; ++i)
        xin[i * 34 + lane + 1] = corr[(b * 3 + i) * NN + n] - corr[(b2 * 3 + i) * NN + n2];
    } else {
#pragma unroll
      for (int i = 0; i < 3; ++i)
        xin[i * 34 + lane + 1] = src[(b * 3 + i) * NN + n]
                               - src_knn[((size_t)pt * KK + lane) * 3 + i];
    }
    if (lane < 3) { xin[lane * 34] = 0.f; xin[lane * 34 + 33] = 0.f; }
    __syncthreads();

    // conv1 (3->64, k=3, pad) + BN + lrelu ; output kpos-major h1t[kpos+1][c]
    float xr[9];
#pragma unroll
    for (int t = 0; t < 9; ++t) xr[t] = xin[(t / 3) * 34 + lane + (t % 3)];
    for (int c = 0; c < 64; ++c) {
      float acc = F[OFF_BE1 + c];
#pragma unroll
      for (int t = 0; t < 9; ++t) acc += F[OFF_W1F + c * 9 + t] * xr[t];
      h1t[(lane + 1) * 64 + c] = (_Float16)lrelu(acc);
    }
    // zero pad rows 0 and 33
    h1t[lane] = (_Float16)0.f;           h1t[32 + lane] = (_Float16)0.f;
    h1t[33 * 64 + lane] = (_Float16)0.f; h1t[33 * 64 + 32 + lane] = (_Float16)0.f;
    __syncthreads();

    // conv2 (64->128, k=3) as WMMA GEMM [128 x 192] x [192 x 32]
    // inner index r = tau*64 + ci; each 32-wide K-slice has constant tau
#pragma unroll 2
    for (int mt = 0; mt < 8; ++mt) {
      v8f c0 = {}; v8f c1 = {};
#pragma unroll
      for (int st = 0; st < 6; ++st) {
        int tau = st >> 1, cbase = (st & 1) << 5;
        v16h af = load_frag(H + HOFF_W2H + (mt * 16 + m) * 192 + st * 32, hh);
        v16h b0 = load_frag(h1t + (m + tau) * 64 + cbase, hh);
        v16h b1 = load_frag(h1t + (16 + m + tau) * 64 + cbase, hh);
        c0 = wmma16(af, b0, c0);
        c1 = wmma16(af, b1, c1);
      }
      // D layout: lane holds col (m), rows o = mt*16 + r + 8*hh (r consecutive)
      v8h* px0 = (v8h*)(x2t + m * 128 + mt * 16 + (hh << 3));
      v8h* px1 = (v8h*)(x2t + (16 + m) * 128 + mt * 16 + (hh << 3));
      v8h o0, o1;
#pragma unroll
      for (int r = 0; r < 8; ++r) {
        int o = mt * 16 + r + (hh << 3);
        o0[r] = (_Float16)lrelu(c0[r] + F[OFF_BE2 + o]);
        o1[r] = (_Float16)lrelu(c1[r] + F[OFF_BE2 + o]);
      }
      if (br == 0) {
        *px0 = o0; *px1 = o1;
      } else {
        v8h p0 = *px0, p1 = *px1;
#pragma unroll
        for (int r = 0; r < 8; ++r) {
          p0[r] = (_Float16)((float)p0[r] - (float)o0[r]);
          p1[r] = (_Float16)((float)p1[r] - (float)o1[r]);
        }
        *px0 = p0; *px1 = p1;
      }
    }
    __syncthreads();
  }

  // MLP m2 -> a2b, m3 -> l3b (first layers via WMMA on x2t)
  mlp16(H + HOFF_M2W1, F + OFF_M2BE1, F + OFF_M2W2, F + OFF_M2BE2, x2t, t1, a2b, lane);
  mlp16(H + HOFF_M3W1, F + OFF_M3BE1, F + OFF_M3W2, F + OFF_M3BE2, x2t, t1, l3b, lane);

  // softmax over kpos + weighted sum -> s[16]
  if (lane < 16) {
    float mx = -3.4e38f;
    for (int k = 0; k < 32; ++k) mx = fmaxf(mx, l3b[lane * 32 + k]);
    float sum = 0.f, sa = 0.f;
    for (int k = 0; k < 32; ++k) {
      float e = expf(l3b[lane * 32 + k] - mx);
      sum += e; sa += a2b[lane * 32 + k] * e;
    }
    sbuf[lane] = sa / sum;
  }
  __syncthreads();
  if (lane < 8) {
    float acc = F[OFF_M4BE1 + lane];
    for (int c = 0; c < 16; ++c) acc += F[OFF_M4W1 + lane * 16 + c] * sbuf[c];
    h4b[lane] = lrelu(acc);
  }
  __syncthreads();
  if (lane == 0) {
    float o = F[OFF_M4B2];
    for (int j = 0; j < 8; ++j) o += F[OFF_M4W2 + j] * h4b[j];
    conf[pt] = 1.f - tanhf(fabsf(o));
  }
  __syncthreads();
}

// ---------------- K3: per-batch csum/src2/corr2/H ---------------------------
__global__ __launch_bounds__(256)
void k_stats(const float* __restrict__ src, const float* __restrict__ tgt,
             const float* __restrict__ conf, float* __restrict__ stat) {
  __shared__ float red[9][256];
  __shared__ float c0[7];
  int b = blockIdx.x, tid = threadIdx.x;
  float a[7] = {0, 0, 0, 0, 0, 0, 0};
  for (int n = tid; n < NN; n += 256) {
    float cf = conf[b * NN + n];
    a[0] += cf;
#pragma unroll
    for (int i = 0; i < 3; ++i) {
      a[1 + i] += src[(b * 3 + i) * NN + n] * cf;
      a[4 + i] += tgt[(b * 3 + i) * NN + n] * cf;
    }
  }
  for (int j = 0; j < 7; ++j) red[j][tid] = a[j];
  __syncthreads();
  for (int s = 128; s > 0; s >>= 1) {
    if (tid < s) for (int j = 0; j < 7; ++j) red[j][tid] += red[j][tid + s];
    __syncthreads();
  }
  if (tid == 0) {
    float cs = red[0][0];
    c0[6] = cs;
    stat[b * 16 + 6] = cs;
    for (int i = 0; i < 3; ++i) {
      c0[i] = red[1 + i][0] / cs;      stat[b * 16 + i] = c0[i];
      c0[3 + i] = red[4 + i][0] / cs;  stat[b * 16 + 3 + i] = c0[3 + i];
    }
  }
  __syncthreads();
  float h[9] = {0, 0, 0, 0, 0, 0, 0, 0, 0};
  for (int n = tid; n < NN; n += 256) {
    float cf = conf[b * NN + n];
    float ds[3], dt[3];
#pragma unroll
    for (int i = 0; i < 3; ++i) {
      ds[i] = (src[(b * 3 + i) * NN + n] - c0[i]) * cf;
      dt[i] = tgt[(b * 3 + i) * NN + n] - c0[3 + i];
    }
#pragma unroll
    for (int i = 0; i < 3; ++i)
#pragma unroll
      for (int j = 0; j < 3; ++j) h[i * 3 + j] += ds[i] * dt[j];
  }
  for (int j = 0; j < 9; ++j) red[j][tid] = h[j];
  __syncthreads();
  for (int s = 128; s > 0; s >>= 1) {
    if (tid < s) for (int j = 0; j < 9; ++j) red[j][tid] += red[j][tid + s];
    __syncthreads();
  }
  if (tid == 0)
    for (int j = 0; j < 9; ++j) stat[b * 16 + 7 + j] = red[j][0];
}

// ---------------- K4: per-batch 3x3 Kabsch SVD -> R,t -----------------------
__global__ void k_svd(const float* __restrict__ stat, float* __restrict__ out) {
  if (threadIdx.x != 0) return;
  int b = blockIdx.x;
  const float* S = stat + b * 16;
  float Hm[3][3], A[3][3];
  for (int i = 0; i < 3; ++i)
    for (int j = 0; j < 3; ++j) Hm[i][j] = S[7 + i * 3 + j];
  for (int i = 0; i < 3; ++i)
    for (int j = 0; j < 3; ++j) {
      float a = 0.f;
      for (int k = 0; k < 3; ++k) a += Hm[k][i] * Hm[k][j];
      A[i][j] = a;
    }
  float V[3][3] = {{1, 0, 0}, {0, 1, 0}, {0, 0, 1}};
  for (int sweep = 0; sweep < 16; ++sweep) {
    const int pi[3] = {0, 0, 1}, qi[3] = {1, 2, 2};
    for (int r = 0; r < 3; ++r) {
      int p = pi[r], q = qi[r];
      float apq = A[p][q];
      if (fabsf(apq) < 1e-20f) continue;
      float tau = (A[q][q] - A[p][p]) / (2.f * apq);
      float t = copysignf(1.f, tau) / (fabsf(tau) + sqrtf(1.f + tau * tau));
      float c = rsqrtf(1.f + t * t);
      float s = t * c;
      for (int k = 0; k < 3; ++k) {
        float akp = A[k][p], akq = A[k][q];
        A[k][p] = c * akp - s * akq; A[k][q] = s * akp + c * akq;
      }
      for (int k = 0; k < 3; ++k) {
        float apk = A[p][k], aqk = A[q][k];
        A[p][k] = c * apk - s * aqk; A[q][k] = s * apk + c * aqk;
      }
      for (int k = 0; k < 3; ++k) {
        float vkp = V[k][p], vkq = V[k][q];
        V[k][p] = c * vkp - s * vkq; V[k][q] = s * vkp + c * vkq;
      }
    }
  }
  float d[3] = {A[0][0], A[1][1], A[2][2]};
  for (int i = 0; i < 2; ++i)
    for (int j = i + 1; j < 3; ++j)
      if (d[j] > d[i]) {
        float td = d[i]; d[i] = d[j]; d[j] = td;
        for (int k = 0; k < 3; ++k) { float tv = V[k][i]; V[k][i] = V[k][j]; V[k][j] = tv; }
      }
  float sv[3];
  for (int i = 0; i < 3; ++i) sv[i] = sqrtf(fmaxf(d[i], 0.f));
  float U[3][3];
  for (int col = 0; col < 3; ++col) {
    float u[3];
    for (int i = 0; i < 3; ++i) {
      float a = 0.f;
      for (int j = 0; j < 3; ++j) a += Hm[i][j] * V[j][col];
      u[i] = a;
    }
    if (sv[col] > 1e-7f * sv[0] + 1e-30f) {
      for (int i = 0; i < 3; ++i) U[i][col] = u[i] / sv[col];
    } else if (col == 2) {
      U[0][2] = U[1][0] * U[2][1] - U[2][0] * U[1][1];
      U[1][2] = U[2][0] * U[0][1] - U[0][0] * U[2][1];
      U[2][2] = U[0][0] * U[1][1] - U[1][0] * U[0][1];
    } else {
      for (int i = 0; i < 3; ++i) U[i][col] = (i == col) ? 1.f : 0.f;
    }
  }
  float R[3][3];
  for (int i = 0; i < 3; ++i)
    for (int j = 0; j < 3; ++j) {
      float a = 0.f;
      for (int k = 0; k < 3; ++k) a += V[i][k] * U[j][k];
      R[i][j] = a;
    }
  float det = R[0][0] * (R[1][1] * R[2][2] - R[1][2] * R[2][1])
            - R[0][1] * (R[1][0] * R[2][2] - R[1][2] * R[2][0])
            + R[0][2] * (R[1][0] * R[2][1] - R[1][1] * R[2][0]);
  if (det < 0.f) {
    for (int k = 0; k < 3; ++k) V[k][2] = -V[k][2];
    for (int i = 0; i < 3; ++i)
      for (int j = 0; j < 3; ++j) {
        float a = 0.f;
        for (int k = 0; k < 3; ++k) a += V[i][k] * U[j][k];
        R[i][j] = a;
      }
  }
  for (int i = 0; i < 3; ++i)
    for (int j = 0; j < 3; ++j) out[b * 9 + i * 3 + j] = R[i][j];
  for (int i = 0; i < 3; ++i) {
    float a = S[3 + i];
    for (int j = 0; j < 3; ++j) a -= R[i][j] * S[j];
    out[72 + b * 3 + i] = a;
  }
}

// ---------------- K5: per-batch top-256 via bitonic sort --------------------
__global__ __launch_bounds__(1024)
void k_topk(const float* __restrict__ conf, int* __restrict__ topk) {
  __shared__ unsigned long long key[1024];
  int b = blockIdx.x, i = threadIdx.x;
  unsigned int fb = __float_as_uint(conf[b * NN + i]); // conf >= 0 -> monotone bits
  key[i] = ((unsigned long long)fb << 32) | (unsigned int)(0xFFFFFFFFu - (unsigned int)i);
  __syncthreads();
  for (int k = 2; k <= 1024; k <<= 1) {
    for (int j = k >> 1; j > 0; j >>= 1) {
      int ixj = i ^ j;
      if (ixj > i) {
        unsigned long long x = key[i], y = key[ixj];
        bool sw = ((i & k) == 0) ? (x < y) : (x > y); // descending overall
        if (sw) { key[i] = y; key[ixj] = x; }
      }
      __syncthreads();
    }
  }
  if (i < PPK)
    topk[b * PPK + i] = (int)(0xFFFFFFFFu - (unsigned int)(key[i] & 0xFFFFFFFFull));
}

// ---------------- K6a: G[b,d,n] = sum_j R[d,j]*src[b,j,n] + t[d] ------------
__global__ __launch_bounds__(256)
void k_transform(const float* __restrict__ src, const float* __restrict__ out,
                 float* __restrict__ G) {
  int tid = blockIdx.x * blockDim.x + threadIdx.x;
  if (tid >= BB * 3 * NN) return;
  int n = tid % NN;
  int d = (tid / NN) % 3;
  int b = tid / (3 * NN);
  float g = out[72 + b * 3 + d];
  for (int j = 0; j < 3; ++j) g += out[b * 9 + d * 3 + j] * src[(b * 3 + j) * NN + n];
  G[(b * 3 + d) * NN + n] = g;
}

// ---------------- K6b: gathers for the four gathered outputs ----------------
__global__ __launch_bounds__(256)
void k_outputs(const float* __restrict__ src, const float* __restrict__ corr,
               const float* __restrict__ G, const int* __restrict__ topk,
               const int* __restrict__ src_idx, float* __restrict__ out) {
  int tid = blockIdx.x * blockDim.x + threadIdx.x;
  if (tid >= BB * 3 * PPK * KK) return;
  int k = tid & 31;
  int p = (tid >> 5) & 255;
  int rem = tid >> 13;        // 32*256
  int d = rem % 3, b = rem / 3;
  int np = topk[b * PPK + p];
  float cp = corr[(b * 3 + d) * NN + np];
  float sp = src[(b * 3 + d) * NN + np];
  float gp = G[(b * 3 + d) * NN + np];
  if (k == 0) {
    out[96 + (b * 3 + d) * PPK + p]   = sp;  // src_kp
    out[6240 + (b * 3 + d) * PPK + p] = cp;  // tgt_kp
  }
  int q = src_idx[(b * NN + np) * KK + k];
  int b2 = q >> 10, n2 = q & 1023;
  // tgt_kp_knn: knn_distance gather (natural [B*N,3] reindex of corr^T)
  out[208992 + ((b * 3 + d) * PPK + p) * KK + k] = cp - corr[(b2 * 3 + d) * NN + n2];
  // src_kp_knn: knn2 gather (raw-flat reindex of transposed src_tr == G flat)
  out[12384 + ((b * 3 + d) * PPK + p) * KK + k] = gp - G[(size_t)q * 3 + d];
}

// ---------------- host launch ----------------------------------------------
extern "C" void kernel_launch(void* const* d_in, const int* in_sizes, int n_in,
                              void* d_out, int out_size, void* d_ws, size_t ws_size,
                              hipStream_t stream) {
  (void)in_sizes; (void)n_in; (void)out_size; (void)ws_size;
  const float* src     = (const float*)d_in[0];
  const float* tgt     = (const float*)d_in[1];
  const float* scores  = (const float*)d_in[2];
  const float* src_knn = (const float*)d_in[3];
  DiscParams dp;
  for (int j = 0; j < NPARAMS; ++j) dp.p[j] = (const float*)d_in[4 + j];
  const int* src_idx = (const int*)d_in[4 + NPARAMS];

  float* ws      = (float*)d_ws;
  float* ws_corr = ws;                 // 24576
  float* ws_conf = ws + 24576;         // 8192
  float* ws_stat = ws + 32768;         // 128
  int*   ws_topk = (int*)(ws + 32896); // 2048 ints
  float* ws_G    = ws + 34944;         // 24576
  float* out = (float*)d_out;

  k_src_corr<<<BB * NN, 128, 0, stream>>>(tgt, scores, ws_corr);
  k_disc<<<(BB * NN) / 4, 128, DISC_LDS_BYTES, stream>>>(dp, src, src_knn, src_idx,
                                                         ws_corr, ws_conf);
  k_stats<<<BB, 256, 0, stream>>>(src, tgt, ws_conf, ws_stat);
  k_svd<<<BB, 32, 0, stream>>>(ws_stat, out);
  k_topk<<<BB, 1024, 0, stream>>>(ws_conf, ws_topk);
  k_transform<<<(BB * 3 * NN + 255) / 256, 256, 0, stream>>>(src, out, ws_G);
  k_outputs<<<(BB * 3 * PPK * KK + 255) / 256, 256, 0, stream>>>(src, ws_corr, ws_G,
                                                                 ws_topk, src_idx, out);
}
